// PartGeoZe_67147518705777
// MI455X (gfx1250) — matrix operand order repaired and checked by reading, not compiled, not run on hardware
//
// MI455X (gfx1250) implementation of the PartGeoZe-style point-cloud pipeline.
// b=2, n=16384 points, s=128 nodes, d=512 dims.
// All dense contractions run on V_WMMA_F32_16X16X32_F16 (f16 operands staged in
// LDS, f32 accumulation). Workspace use: ~42 MB of d_ws.
#include <hip/hip_runtime.h>
#include <math.h>

#define NB 2
#define NP 16384
#define NS 128
#define ND 512
#define NFD 33
#define NX 40          // [xyz(3) | normals(3) | fpfh(33) | ones(1)]
#define N_ITERS 30
#define TAU 0.1f
#define W_NORM 1.0f
#define W_FPFH 0.1f
#define SK_EPS 0.01f
#define SK_ITERS 10
#define RPM_ITERS 5
#define DEC_K 8
#define LOGP_M (-9.70406053f)   // -ln(16384)
#define LOGQ_M (-4.85203026f)   // -ln(128)

typedef _Float16 v2h  __attribute__((ext_vector_type(2)));
typedef _Float16 v8h  __attribute__((ext_vector_type(8)));
typedef _Float16 v16h __attribute__((ext_vector_type(16)));
typedef float    v8f  __attribute__((ext_vector_type(8)));

// ---------------------------------------------------------------------------
// WMMA GEMM: C[M,N] = post * (alpha * A(M,K) x B(K,N) + Cin), arbitrary strides.
// Tile 128x128 per 256-thread block (8 wave32 waves); K chunks of 32 staged in
// LDS as f16. Optional K-split (atomicAdd f32 epilogue, C pre-zeroed by caller).
// ALIGNED: M%128==0 && N%128==0 && K%32==0 -> no guards, affine-offset staging.
// BNFAST : sBn==1 -> n-fast staging mapping for coalesced B reads.
// ---------------------------------------------------------------------------
#define TM 128
#define TN 128
#define TK 32
#define LDT 40   // padded row stride in halves (80B, keeps 16B alignment)

template<bool ALIGNED, bool BNFAST>
__launch_bounds__(256)
__global__ void k_wmma_gemm(const float* __restrict__ A, long sAm, long sAk, long sAb,
                            const float* __restrict__ B, long sBk, long sBn, long sBb,
                            const float* __restrict__ Cin, long sCinB,
                            float* __restrict__ C, long sCB,
                            int M, int N, int K, int kSplit,
                            float alpha, float post)
{
    __shared__ __align__(16) _Float16 As[TM * LDT];
    __shared__ __align__(16) _Float16 Bs[TN * LDT];

    const int b  = blockIdx.z / kSplit;
    const int ks = blockIdx.z % kSplit;
    A += (long)b * sAb;
    B += (long)b * sBb;
    C += (long)b * sCB;
    if (Cin) Cin += (long)b * sCinB;

    const int m0   = blockIdx.x * TM;
    const int n0   = blockIdx.y * TN;
    const int tid  = threadIdx.x;
    const int lane = tid & 31;
    const int wid  = tid >> 5;
    const int mw   = m0 + wid * 16;          // this wave's 16 output rows

    const int KC  = (K + TK - 1) / TK;
    const int cps = (KC + kSplit - 1) / kSplit;
    const int c0  = ks * cps;
    int c1 = c0 + cps; if (c1 > KC) c1 = KC;

    // per-thread staging invariants (k-fast A mapping; B mapping by BNFAST)
    const int ar  = (tid * 2) >> 5;              // A tile row
    const int akk = (tid * 2) & 31;              // A k-in-chunk
    const long aoff0 = (long)(m0 + ar) * sAm + (long)akk * sAk;
    int br, bkk;
    if (BNFAST) { br = tid & 127; bkk = (tid >> 7) * 2; }
    else        { br = (tid * 2) >> 5; bkk = (tid * 2) & 31; }
    const long boff0 = (long)(n0 + br) * sBn + (long)bkk * sBk;
    const long bstep = BNFAST ? 4 * sBk : 16 * sBn;   // per-i element step

    v8f acc[8];
    #pragma unroll
    for (int t = 0; t < 8; ++t)
        #pragma unroll
        for (int r = 0; r < 8; ++r) acc[t][r] = 0.0f;

    for (int ci = c0; ci < c1; ++ci) {
        const int k0 = ci * TK;

        // ---- stage A tile (128 rows x 32 K), 2 K-elements per thread-step ----
        if (ALIGNED) {
            const float* ap = A + aoff0 + (long)k0 * sAk;
            float v0[8], v1[8];
            if (sAk == 1) {            // K-contiguous: pair is one b64 load
                #pragma unroll
                for (int i = 0; i < 8; ++i) {
                    float2 p = *(const float2*)(ap + (long)(i * 16) * sAm);
                    v0[i] = p.x; v1[i] = p.y;
                }
            } else {
                #pragma unroll
                for (int i = 0; i < 8; ++i) {
                    v0[i] = ap[(long)(i * 16) * sAm];
                    v1[i] = ap[(long)(i * 16) * sAm + sAk];
                }
            }
            #pragma unroll
            for (int i = 0; i < 8; ++i) {
                int e = (tid + i * 256) * 2;
                int r = e >> 5, kk = e & 31;
                v2h pv; pv[0] = (_Float16)v0[i]; pv[1] = (_Float16)v1[i];
                *(v2h*)&As[r * LDT + kk] = pv;
            }
        } else {
            float v0[8], v1[8];
            #pragma unroll
            for (int i = 0; i < 8; ++i) {
                int e  = (tid + i * 256) * 2;
                long m = m0 + (e >> 5);
                long k = k0 + (e & 31);
                long mc  = m < M ? m : (long)(M - 1);
                long kc0 = k < K ? k : (long)(K - 1);
                long kc1 = (k + 1) < K ? (k + 1) : (long)(K - 1);
                v0[i] = A[mc * sAm + kc0 * sAk];
                v1[i] = A[mc * sAm + kc1 * sAk];
            }
            #pragma unroll
            for (int i = 0; i < 8; ++i) {
                int e  = (tid + i * 256) * 2;
                int r  = e >> 5, kk = e & 31;
                float a0 = v0[i], a1 = v1[i];
                long m = m0 + r, k = k0 + kk;
                if (m >= M || k >= K) a0 = 0.0f;
                if (m >= M || (k + 1) >= K) a1 = 0.0f;
                v2h pv; pv[0] = (_Float16)a0; pv[1] = (_Float16)a1;
                *(v2h*)&As[r * LDT + kk] = pv;
            }
        }
        // ---- stage B tile (128 cols x 32 K) ----
        if (ALIGNED) {
            const float* bp = B + boff0 + (long)k0 * sBk;
            float v0[8], v1[8];
            if (sBk == 1) {            // K-contiguous: pair is one b64 load
                #pragma unroll
                for (int i = 0; i < 8; ++i) {
                    float2 p = *(const float2*)(bp + (long)i * bstep);
                    v0[i] = p.x; v1[i] = p.y;
                }
            } else {
                #pragma unroll
                for (int i = 0; i < 8; ++i) {
                    v0[i] = bp[(long)i * bstep];
                    v1[i] = bp[(long)i * bstep + sBk];
                }
            }
            #pragma unroll
            for (int i = 0; i < 8; ++i) {
                int r, kk;
                if (BNFAST) { r = br; kk = bkk + i * 4; }
                else        { int e = (tid + i * 256) * 2; r = e >> 5; kk = e & 31; }
                v2h pv; pv[0] = (_Float16)v0[i]; pv[1] = (_Float16)v1[i];
                *(v2h*)&Bs[r * LDT + kk] = pv;
            }
        } else {
            float v0[8], v1[8];
            int rr[8], kk8[8];
            #pragma unroll
            for (int i = 0; i < 8; ++i) {
                int idx = tid + i * 256;       // 0..2047 pairs
                int r, kk;
                if (BNFAST) { r = idx & 127; kk = (idx >> 7) * 2; }
                else        { int e = idx * 2; r = e >> 5; kk = e & 31; }
                rr[i] = r; kk8[i] = kk;
                long n = n0 + r;
                long k = k0 + kk;
                long nc  = n < N ? n : (long)(N - 1);
                long kc0 = k < K ? k : (long)(K - 1);
                long kc1 = (k + 1) < K ? (k + 1) : (long)(K - 1);
                v0[i] = B[kc0 * sBk + nc * sBn];
                v1[i] = B[kc1 * sBk + nc * sBn];
            }
            #pragma unroll
            for (int i = 0; i < 8; ++i) {
                int r = rr[i], kk = kk8[i];
                float b0 = v0[i], b1 = v1[i];
                long n = n0 + r, k = k0 + kk;
                if (n >= N || k >= K) b0 = 0.0f;
                if (n >= N || (k + 1) >= K) b1 = 0.0f;
                v2h pv; pv[0] = (_Float16)b0; pv[1] = (_Float16)b1;
                *(v2h*)&Bs[r * LDT + kk] = pv;
            }
        }
        __syncthreads();

        // prefetch a token of next A chunk into L2
        if (ci + 1 < c1)
            __builtin_prefetch(&A[aoff0 + (long)(ci + 1) * TK * sAk], 0, 0);

        // ---- A fragment (16x32 f16): lanes 0-15 & 16-31 both hold M=0..15 ----
        const int am = (mw - m0) + (lane & 15);
        const int kh = (lane >> 4) * 8;
        v16h af;
        {
            v8h lo = *(const v8h*)&As[am * LDT + kh];
            v8h hi = *(const v8h*)&As[am * LDT + kh + 16];
            #pragma unroll
            for (int i = 0; i < 8; ++i) { af[i] = lo[i]; af[8 + i] = hi[i]; }
        }
        // ---- 8 WMMAs across the 128-wide N tile, B fragment double-buffered ----
        const int kb   = (lane >> 4) * 8;
        const int bcol = lane & 15;
        auto loadB = [&](int nt) -> v16h {
            v16h bf;
            v8h lo = *(const v8h*)&Bs[(nt * 16 + bcol) * LDT + kb];
            v8h hi = *(const v8h*)&Bs[(nt * 16 + bcol) * LDT + kb + 16];
            #pragma unroll
            for (int i = 0; i < 8; ++i) { bf[i] = lo[i]; bf[8 + i] = hi[i]; }
            return bf;
        };
        v16h bcur = loadB(0);
        #pragma unroll
        for (int nt = 0; nt < 8; ++nt) {
            v16h bnext = bcur;
            if (nt < 7) bnext = loadB(nt + 1);
            acc[nt] = __builtin_amdgcn_wmma_f32_16x16x32_f16(
                false, af, false, bcur, (short)0, acc[nt], false, false);
            bcur = bnext;
        }
        __syncthreads();
    }

    // ---- epilogue (C/D layout: lanes 0-15 M=r, lanes 16-31 M=8+r) ----
    const int mrow = mw + (lane >> 4) * 8;
    #pragma unroll
    for (int nt = 0; nt < 8; ++nt) {
        const int col = n0 + nt * 16 + (lane & 15);
        if (!ALIGNED && col >= N) continue;
        #pragma unroll
        for (int r = 0; r < 8; ++r) {
            int m = mrow + r;
            if (!ALIGNED && m >= M) continue;
            float v = alpha * acc[nt][r];
            long off = (long)m * N + col;
            if (kSplit > 1) {
                atomicAdd(&C[off], v);
            } else {
                if (Cin) v += Cin[off];
                C[off] = post * v;
            }
        }
    }
}

// ---------------------------------------------------------------------------
// Farthest point sampling: 1 block per batch, 128 sequential argmax rounds.
// ---------------------------------------------------------------------------
__launch_bounds__(1024)
__global__ void k_fps(const float* __restrict__ xyz, int* __restrict__ idx)
{
    __shared__ float sval[1024];
    __shared__ int   sidx[1024];
    __shared__ int   sfar;
    const int b   = blockIdx.x;
    const int tid = threadIdx.x;
    const float* X = xyz + (long)b * NP * 3;
    float dl[16];
    #pragma unroll
    for (int j = 0; j < 16; ++j) dl[j] = 1e10f;
    int far = 0;
    for (int it = 0; it < NS; ++it) {
        if (tid == 0) idx[b * NS + it] = far;
        float cx = X[far * 3 + 0], cy = X[far * 3 + 1], cz = X[far * 3 + 2];
        float bv = -1.0f; int bi = 0;
        #pragma unroll
        for (int j = 0; j < 16; ++j) {
            int p = j * 1024 + tid;
            float dx = X[p * 3 + 0] - cx, dy = X[p * 3 + 1] - cy, dz = X[p * 3 + 2] - cz;
            float d = dx * dx + dy * dy + dz * dz;
            dl[j] = fminf(dl[j], d);
            if (dl[j] > bv) { bv = dl[j]; bi = p; }
        }
        sval[tid] = bv; sidx[tid] = bi;
        __syncthreads();
        for (int s = 512; s > 0; s >>= 1) {
            if (tid < s) {
                float ov = sval[tid + s]; int oi = sidx[tid + s];
                if (ov > sval[tid] || (ov == sval[tid] && oi < sidx[tid])) {
                    sval[tid] = ov; sidx[tid] = oi;
                }
            }
            __syncthreads();
        }
        if (tid == 0) sfar = sidx[0];
        __syncthreads();
        far = sfar;
    }
}

__global__ void k_fill(float* p, float v, long n) {
    long t = (long)blockIdx.x * blockDim.x + threadIdx.x;
    if (t < n) p[t] = v;
}

__global__ void k_gather(const float* xyz, const float* normals, const float* fpfhs,
                         const int* idx, float* proto, float* cnorm, float* cfpfh)
{
    int t = blockIdx.x * blockDim.x + threadIdx.x;
    if (t >= NB * NS) return;
    int b = t / NS;
    int i = idx[t];
    const float* X  = xyz     + ((long)b * NP + i) * 3;
    const float* Nn = normals + ((long)b * NP + i) * 3;
    const float* F  = fpfhs   + ((long)b * NP + i) * NFD;
    for (int c = 0; c < 3; ++c) { proto[t * 3 + c] = X[c]; cnorm[t * 3 + c] = Nn[c]; }
    for (int c = 0; c < NFD; ++c) cfpfh[t * NFD + c] = F[c];
}

// Build X40 = [xyz | normals | fpfh | 1] once (B operand of the cluster GEMM).
__global__ void k_x40(const float* xyz, const float* normals, const float* fpfhs, float* x40)
{
    int t = blockIdx.x * blockDim.x + threadIdx.x;
    if (t >= NB * NP) return;
    const float* X  = xyz     + (long)t * 3;
    const float* Nn = normals + (long)t * 3;
    const float* F  = fpfhs   + (long)t * NFD;
    float* o = x40 + (long)t * NX;
    o[0] = X[0]; o[1] = X[1]; o[2] = X[2];
    o[3] = Nn[0]; o[4] = Nn[1]; o[5] = Nn[2];
    for (int c = 0; c < NFD; ++c) o[6 + c] = F[c];
    o[39] = 1.0f;
}

__device__ __forceinline__ float cluster_cost(const float* q, float px, float py, float pz,
                                              float nx, float ny, float nz, const float* f)
{
    float dx = px - q[0], dy = py - q[1], dz = pz - q[2];
    float c = sqrtf(fmaxf(dx * dx + dy * dy + dz * dz, 1e-12f));
    c += W_NORM * (1.0f - (nx * q[3] + ny * q[4] + nz * q[5]));
    float fd = 0.0f;
    #pragma unroll
    for (int i = 0; i < NFD; ++i) { float tt = f[i] - q[6 + i]; fd += tt * tt; }
    c += W_FPFH * sqrtf(fmaxf(fd, 1e-12f));
    return c;
}

// Soft assignment gamma (softmax over 128 protos), written s-major (gammaT).
__launch_bounds__(256)
__global__ void k_cluster(const float* __restrict__ xyz, const float* __restrict__ normals,
                          const float* __restrict__ fpfhs,
                          const float* __restrict__ proto, const float* __restrict__ cnorm,
                          const float* __restrict__ cfpfh, float* __restrict__ gammaT)
{
    __shared__ float sp[NS * 39];
    const int b = blockIdx.y;
    const int tid = threadIdx.x;
    for (int i = tid; i < NS * 39; i += 256) {
        int s = i / 39, c = i % 39;
        float v;
        if (c < 3)      v = proto[(b * NS + s) * 3 + c];
        else if (c < 6) v = cnorm[(b * NS + s) * 3 + (c - 3)];
        else            v = cfpfh[(b * NS + s) * NFD + (c - 6)];
        sp[i] = v;
    }
    __syncthreads();
    if (tid < NS) {  // l2-normalize the proto normals in LDS
        float a = sp[tid * 39 + 3], bb = sp[tid * 39 + 4], cc = sp[tid * 39 + 5];
        float nrm = fmaxf(sqrtf(a * a + bb * bb + cc * cc), 1e-12f);
        sp[tid * 39 + 3] = a / nrm; sp[tid * 39 + 4] = bb / nrm; sp[tid * 39 + 5] = cc / nrm;
    }
    __syncthreads();

    const int p = blockIdx.x * 256 + tid;
    const float* Xp = xyz + ((long)b * NP + p) * 3;
    float px = Xp[0], py = Xp[1], pz = Xp[2];
    const float* Npn = normals + ((long)b * NP + p) * 3;
    float nx = Npn[0], ny = Npn[1], nz = Npn[2];
    float nn = fmaxf(sqrtf(nx * nx + ny * ny + nz * nz), 1e-12f);
    nx /= nn; ny /= nn; nz /= nn;
    float f[NFD];
    const float* Fp = fpfhs + ((long)b * NP + p) * NFD;
    #pragma unroll
    for (int i = 0; i < NFD; ++i) f[i] = Fp[i];

    float mx = -1e30f;
    for (int s = 0; s < NS; ++s) {
        float lg = -cluster_cost(&sp[s * 39], px, py, pz, nx, ny, nz, f) * (1.0f / TAU);
        mx = fmaxf(mx, lg);
    }
    float sum = 0.0f;
    for (int s = 0; s < NS; ++s) {
        float lg = -cluster_cost(&sp[s * 39], px, py, pz, nx, ny, nz, f) * (1.0f / TAU);
        sum += __expf(lg - mx);
    }
    float inv = 1.0f / sum;
    float* gout = gammaT + (long)b * NS * NP + p;
    for (int s = 0; s < NS; ++s) {
        float lg = -cluster_cost(&sp[s * 39], px, py, pz, nx, ny, nz, f) * (1.0f / TAU);
        gout[(long)s * NP] = __expf(lg - mx) * inv;
    }
}

// protos/cnorm/cfpfh = C40[:, 0:39] / C40[:, 39]
__global__ void k_normalize(const float* C40, float* proto, float* cnorm, float* cfpfh,
                            int writeNorm)
{
    int t = blockIdx.x * blockDim.x + threadIdx.x;
    if (t >= NB * NS) return;
    const float* a = C40 + (long)t * NX;
    float inv = 1.0f / (a[39] + 1e-6f);
    for (int c = 0; c < 3; ++c) proto[t * 3 + c] = a[c] * inv;
    if (writeNorm) for (int c = 0; c < 3; ++c) cnorm[t * 3 + c] = a[3 + c] * inv;
    for (int c = 0; c < NFD; ++c) cfpfh[t * NFD + c] = a[6 + c] * inv;
}

__global__ void k_div_rows(float* X, const float* C40, int cols) {
    long t = (long)blockIdx.x * blockDim.x + threadIdx.x;
    if (t >= (long)NB * NS * cols) return;
    long row = t / cols;
    X[t] /= (C40[row * NX + 39] + 1e-6f);
}

__global__ void k_rownorms(const float* X, float* out, int cols) {
    int t = blockIdx.x * blockDim.x + threadIdx.x;
    if (t >= NB * NS) return;
    const float* r = X + (long)t * cols;
    float s = 0.0f;
    for (int c = 0; c < cols; ++c) s += r[c] * r[c];
    out[t] = fmaxf(sqrtf(s), 1e-12f);
}

__global__ void k_dmat(const float* cfpfh, const float* nrm, float* D) {
    int t = blockIdx.x * blockDim.x + threadIdx.x;
    if (t >= NB * NS * NS) return;
    int b = t / (NS * NS); int ij = t % (NS * NS); int i = ij / NS, j = ij % NS;
    const float* a = cfpfh + ((long)b * NS + i) * NFD;
    const float* c = cfpfh + ((long)b * NS + j) * NFD;
    float d = 0.0f;
    #pragma unroll
    for (int k = 0; k < NFD; ++k) d += a[k] * c[k];
    d /= (nrm[b * NS + i] * nrm[b * NS + j]);
    d = fmaxf(d, 0.001f);
    D[t] = d / 0.01f;
}

__global__ void k_fmat(const float* nf, const float* nrm, float* F) {
    int t = blockIdx.x * blockDim.x + threadIdx.x;
    if (t >= NB * NS * NS) return;
    int b = t / (NS * NS); int ij = t % (NS * NS); int i = ij / NS, j = ij % NS;
    const float* a = nf + ((long)b * NS + i) * ND;
    const float* c = nf + ((long)b * NS + j) * ND;
    float d = 0.0f;
    for (int k = 0; k < ND; ++k) d += a[k] * c[k];
    d /= (nrm[b * NS + i] * nrm[b * NS + j]);
    d = fminf(fmaxf(d, 0.01f), 0.9f);
    float v = (i == j ? 0.0f : d) + 1.0f / ((float)NS * NS + 1e-6f);
    F[t] = v / 0.1f;
}

__global__ void k_rpm_row(float* A) {
    int t = blockIdx.x * blockDim.x + threadIdx.x;
    if (t >= NB * NS) return;
    float* r = A + (long)t * NS;
    float mx = -1e30f;
    for (int j = 0; j < NS; ++j) mx = fmaxf(mx, r[j]);
    float s = 0.0f;
    for (int j = 0; j < NS; ++j) s += __expf(r[j] - mx);
    float l = mx + logf(s);
    for (int j = 0; j < NS; ++j) r[j] -= l;
}

__global__ void k_rpm_col(float* A) {
    int t = blockIdx.x * blockDim.x + threadIdx.x;
    if (t >= NB * NS) return;
    int b = t / NS, j = t % NS;
    float* base = A + (long)b * NS * NS + j;
    float mx = -1e30f;
    for (int i = 0; i < NS; ++i) mx = fmaxf(mx, base[i * NS]);
    float s = 0.0f;
    for (int i = 0; i < NS; ++i) s += __expf(base[i * NS] - mx);
    float l = mx + logf(s);
    for (int i = 0; i < NS; ++i) base[i * NS] -= l;
}

// attn = rownorm(exp(D_rpm + F_rpm)), stored into D
__global__ void k_gattn_attn(float* D, const float* F) {
    int t = blockIdx.x * blockDim.x + threadIdx.x;
    if (t >= NB * NS) return;
    float* d = D + (long)t * NS;
    const float* f = F + (long)t * NS;
    float s = 0.0f;
    for (int j = 0; j < NS; ++j) { float v = __expf(d[j] + f[j]); d[j] = v; s += v; }
    float inv = 1.0f / s;
    for (int j = 0; j < NS; ++j) d[j] *= inv;
}

__global__ void k_s2p_cost(const float* xyz, const float* proto, float* M) {
    long t = (long)blockIdx.x * blockDim.x + threadIdx.x;
    if (t >= (long)NB * NP) return;
    int b = t / NP;
    const float* X = xyz + t * 3;
    float px = X[0], py = X[1], pz = X[2];
    float* row = M + t * NS;
    const float* P = proto + (long)b * NS * 3;
    for (int s = 0; s < NS; ++s) {
        float dx = px - P[s * 3 + 0], dy = py - P[s * 3 + 1], dz = pz - P[s * 3 + 2];
        float d = sqrtf(fmaxf(dx * dx + dy * dy + dz * dz, 1e-12f));
        row[s] = -d * (1.0f / SK_EPS);
    }
}

__global__ void k_sk_row(const float* M, const float* v, float* u) {
    long t = (long)blockIdx.x * blockDim.x + threadIdx.x;
    if (t >= (long)NB * NP) return;
    int b = t / NP;
    const float* row = M + t * NS;
    const float* vv = v + b * NS;
    float mx = -1e30f;
    for (int s = 0; s < NS; ++s) mx = fmaxf(mx, row[s] + vv[s]);
    float sm = 0.0f;
    for (int s = 0; s < NS; ++s) sm += __expf(row[s] + vv[s] - mx);
    u[t] = LOGP_M - (mx + logf(sm));
}

__launch_bounds__(256)
__global__ void k_sk_col(const float* M, const float* u, float* v) {
    __shared__ float red[256];
    int bs = blockIdx.x;                 // 0 .. NB*NS-1
    int b = bs / NS, s = bs % NS;
    const float* base = M + (long)b * NP * NS + s;
    const float* uu = u + (long)b * NP;
    int tid = threadIdx.x;
    float mx = -1e30f;
    for (int n = tid; n < NP; n += 256) mx = fmaxf(mx, base[(long)n * NS] + uu[n]);
    red[tid] = mx; __syncthreads();
    for (int k = 128; k > 0; k >>= 1) {
        if (tid < k) red[tid] = fmaxf(red[tid], red[tid + k]);
        __syncthreads();
    }
    mx = red[0]; __syncthreads();
    float sm = 0.0f;
    for (int n = tid; n < NP; n += 256) sm += __expf(base[(long)n * NS] + uu[n] - mx);
    red[tid] = sm; __syncthreads();
    for (int k = 128; k > 0; k >>= 1) {
        if (tid < k) red[tid] += red[tid + k];
        __syncthreads();
    }
    if (tid == 0) v[bs] = LOGQ_M - (mx + logf(red[0]));
}

__global__ void k_dscore(float* M, const float* u, const float* v) {
    long t = (long)blockIdx.x * blockDim.x + threadIdx.x;
    if (t >= (long)NB * NP * NS) return;
    long row = t / NS; int s = t % NS; int b = row / NP;
    M[t] = __expf(M[t] + u[row] + v[b * NS + s]);
}

__global__ void k_row_aux(const float* Fm, float* aux) {
    long t = (long)blockIdx.x * blockDim.x + threadIdx.x;
    if (t >= (long)NB * NP) return;
    const float* r = Fm + t * NS;
    float mx = -1e30f;
    for (int s = 0; s < NS; ++s) mx = fmaxf(mx, r[s]);
    float sm = 0.0f;
    for (int s = 0; s < NS; ++s) sm += __expf(r[s] - mx);
    aux[2 * t] = mx; aux[2 * t + 1] = sm;
}

__launch_bounds__(256)
__global__ void k_col_aux(const float* Fm, float* caux) {
    __shared__ float red[256];
    int bs = blockIdx.x;
    int b = bs / NS, s = bs % NS;
    const float* base = Fm + (long)b * NP * NS + s;
    int tid = threadIdx.x;
    float mx = -1e30f;
    for (int n = tid; n < NP; n += 256) mx = fmaxf(mx, base[(long)n * NS]);
    red[tid] = mx; __syncthreads();
    for (int k = 128; k > 0; k >>= 1) {
        if (tid < k) red[tid] = fmaxf(red[tid], red[tid + k]);
        __syncthreads();
    }
    mx = red[0]; __syncthreads();
    float sm = 0.0f;
    for (int n = tid; n < NP; n += 256) sm += __expf(base[(long)n * NS] - mx);
    red[tid] = sm; __syncthreads();
    for (int k = 128; k > 0; k >>= 1) {
        if (tid < k) red[tid] += red[tid + k];
        __syncthreads();
    }
    if (tid == 0) { caux[2 * bs] = mx; caux[2 * bs + 1] = red[0]; }
}

__global__ void k_fs(float* Fm, const float* dsc, const float* raux, const float* caux) {
    long t = (long)blockIdx.x * blockDim.x + threadIdx.x;
    if (t >= (long)NB * NP * NS) return;
    long row = t / NS; int s = t % NS; int b = row / NP;
    float x = Fm[t];
    float fr = __expf(x - raux[2 * row]) / raux[2 * row + 1];
    float fc = __expf(x - caux[2 * (b * NS + s)]) / caux[2 * (b * NS + s) + 1];
    Fm[t] = dsc[t] * fr * fc;
}

__global__ void k_rownorm_ip(float* A, int cols) {
    long t = (long)blockIdx.x * blockDim.x + threadIdx.x;
    if (t >= (long)NB * NP) return;
    float* r = A + t * cols;
    float s = 0.0f;
    for (int j = 0; j < cols; ++j) s += r[j];
    float inv = 1.0f / (s + 1e-20f);
    for (int j = 0; j < cols; ++j) r[j] *= inv;
}

__global__ void k_top8(const float* xyz, const float* normals, const float* proto,
                       const float* cnorm, float* W8, int* I8)
{
    long t = (long)blockIdx.x * blockDim.x + threadIdx.x;
    if (t >= (long)NB * NP) return;
    int b = t / NP;
    const float* X = xyz + t * 3;
    float px = X[0], py = X[1], pz = X[2];
    const float* Nn = normals + t * 3;
    float nx = Nn[0], ny = Nn[1], nz = Nn[2];
    float nn = fmaxf(sqrtf(nx * nx + ny * ny + nz * nz), 1e-12f);
    nx /= nn; ny /= nn; nz /= nn;

    float bd[DEC_K]; int bi[DEC_K];
    #pragma unroll
    for (int k = 0; k < DEC_K; ++k) { bd[k] = 1e30f; bi[k] = 0; }
    const float* P = proto + (long)b * NS * 3;
    for (int s = 0; s < NS; ++s) {
        float dx = px - P[s * 3 + 0], dy = py - P[s * 3 + 1], dz = pz - P[s * 3 + 2];
        float d = sqrtf(fmaxf(dx * dx + dy * dy + dz * dz, 1e-12f));
        if (d < bd[DEC_K - 1]) {
            int k = DEC_K - 1;
            while (k > 0 && bd[k - 1] > d) { bd[k] = bd[k - 1]; bi[k] = bi[k - 1]; --k; }
            bd[k] = d; bi[k] = s;
        }
    }
    float w[DEC_K]; float sum = 0.0f;
    #pragma unroll
    for (int k = 0; k < DEC_K; ++k) {
        const float* mm = cnorm + ((long)b * NS + bi[k]) * 3;
        float nl = fmaxf(sqrtf(mm[0] * mm[0] + mm[1] * mm[1] + mm[2] * mm[2]), 1e-12f);
        float cs = (nx * mm[0] + ny * mm[1] + nz * mm[2]) / nl;
        cs = fmaxf(cs, 0.0f);
        w[k] = (1.0f / (bd[k] + 1e-8f)) * (cs + 0.001f);
        sum += w[k];
    }
    #pragma unroll
    for (int k = 0; k < DEC_K; ++k) { W8[t * 8 + k] = w[k] / sum; I8[t * 8 + k] = bi[k]; }
}

__global__ void k_interp(float* outF, const float* nodeF, const float* W8, const int* I8) {
    long t = (long)blockIdx.x * blockDim.x + threadIdx.x;
    if (t >= (long)NB * NP * ND) return;
    int d = t % ND; long row = t / ND; int b = row / NP;
    const float* nf = nodeF + (long)b * NS * ND;
    float acc = 0.0f;
    #pragma unroll
    for (int k = 0; k < DEC_K; ++k)
        acc += W8[row * 8 + k] * nf[(long)I8[row * 8 + k] * ND + d];
    outF[t] = (outF[t] + acc) * 0.5f;
}

__global__ void k_idx_out(const int* idx, float* out) {
    int t = blockIdx.x * blockDim.x + threadIdx.x;
    if (t >= NB * NS) return;
    out[t] = (float)idx[t];
}

// ---------------------------------------------------------------------------
static void gemm(hipStream_t st,
                 const float* A, long sAm, long sAk, long sAb,
                 const float* B, long sBk, long sBn, long sBb,
                 const float* Cin, long sCinB, float* C, long sCB,
                 int M, int N, int K, int kSplit, float alpha, float post)
{
    dim3 g((M + TM - 1) / TM, (N + TN - 1) / TN, NB * kSplit);
    bool aligned = (M % TM == 0) && (N % TN == 0) && (K % TK == 0);
    bool bnfast  = (sBn == 1);
    if (aligned && bnfast)
        k_wmma_gemm<true, true><<<g, 256, 0, st>>>(A, sAm, sAk, sAb, B, sBk, sBn, sBb,
                                                   Cin, sCinB, C, sCB, M, N, K, kSplit, alpha, post);
    else if (aligned)
        k_wmma_gemm<true, false><<<g, 256, 0, st>>>(A, sAm, sAk, sAb, B, sBk, sBn, sBb,
                                                    Cin, sCinB, C, sCB, M, N, K, kSplit, alpha, post);
    else if (bnfast)
        k_wmma_gemm<false, true><<<g, 256, 0, st>>>(A, sAm, sAk, sAb, B, sBk, sBn, sBb,
                                                    Cin, sCinB, C, sCB, M, N, K, kSplit, alpha, post);
    else
        k_wmma_gemm<false, false><<<g, 256, 0, st>>>(A, sAm, sAk, sAb, B, sBk, sBn, sBb,
                                                     Cin, sCinB, C, sCB, M, N, K, kSplit, alpha, post);
}

extern "C" void kernel_launch(void* const* d_in, const int* in_sizes, int n_in,
                              void* d_out, int out_size, void* d_ws, size_t ws_size,
                              hipStream_t stream)
{
    const float* xyz     = (const float*)d_in[0];
    const float* normals = (const float*)d_in[1];
    const float* fpfhs   = (const float*)d_in[2];
    const float* feats   = (const float*)d_in[3];
    // d_in[4] (text_feats) is unused by the reference.

    float* ws = (float*)d_ws;
    size_t off = 0;
    int*   IDX   = (int*)(ws + off); off += 256;
    float* PROTO = ws + off; off += (size_t)NB * NS * 3;
    float* CNORM = ws + off; off += (size_t)NB * NS * 3;
    float* CFPFH = ws + off; off += (size_t)NB * NS * NFD;
    float* C40   = ws + off; off += (size_t)NB * NS * NX;
    float* X40B  = ws + off; off += (size_t)NB * NP * NX;
    float* GAMT  = ws + off; off += (size_t)NB * NS * NP;   // reused as FBUF later
    float* NFRAW = ws + off; off += (size_t)NB * NS * ND;
    float* RN1   = ws + off; off += (size_t)NB * NS;
    float* RN2   = ws + off; off += (size_t)NB * NS;
    float* DMAT  = ws + off; off += (size_t)NB * NS * NS;
    float* FMAT  = ws + off; off += (size_t)NB * NS * NS;
    float* MMAT  = ws + off; off += (size_t)NB * NP * NS;
    float* UU    = ws + off; off += (size_t)NB * NP;
    float* VV    = ws + off; off += (size_t)NB * NS;
    float* RAUX  = ws + off; off += (size_t)NB * NP * 2;
    float* CAUX  = ws + off; off += (size_t)NB * NS * 2;
    float* W8    = ws + off; off += (size_t)NB * NP * 8;
    int*   I8    = (int*)(ws + off); off += (size_t)NB * NP * 8;
    float* FBUF  = GAMT;   // gamma no longer needed once node aggregation is done

    float* outF    = (float*)d_out;                    // feats (b,n,512)
    float* outNode = outF + (size_t)NB * NP * ND;      // node_feats (b,128,512)
    float* outIdx  = outNode + (size_t)NB * NS * ND;   // idx (b,128) as float

    // ---- FPS + initial prototypes ----
    k_fps<<<NB, 1024, 0, stream>>>(xyz, IDX);
    k_gather<<<1, 256, 0, stream>>>(xyz, normals, fpfhs, IDX, PROTO, CNORM, CFPFH);
    k_x40<<<(NB * NP + 255) / 256, 256, 0, stream>>>(xyz, normals, fpfhs, X40B);

    // ---- clustering: gamma kernel + one WMMA GEMM (gammaT x X40) per iter ----
    for (int it = 0; it < N_ITERS; ++it) {
        k_cluster<<<dim3(NP / 256, NB), 256, 0, stream>>>(
            xyz, normals, fpfhs, PROTO, CNORM, CFPFH, GAMT);
        k_fill<<<(NB * NS * NX + 255) / 256, 256, 0, stream>>>(C40, 0.0f, NB * NS * NX);
        gemm(stream, GAMT, NP, 1, (long)NS * NP, X40B, NX, 1, (long)NP * NX,
             nullptr, 0, C40, (long)NS * NX, NS, NX, NP, 8, 1.0f, 1.0f);
        k_normalize<<<1, 256, 0, stream>>>(C40, PROTO, CNORM, CFPFH, 1);
    }
    // final gamma with converged protos; re-derive protos/fpfh prototypes
    k_cluster<<<dim3(NP / 256, NB), 256, 0, stream>>>(
        xyz, normals, fpfhs, PROTO, CNORM, CFPFH, GAMT);
    k_fill<<<(NB * NS * NX + 255) / 256, 256, 0, stream>>>(C40, 0.0f, NB * NS * NX);
    gemm(stream, GAMT, NP, 1, (long)NS * NP, X40B, NX, 1, (long)NP * NX,
         nullptr, 0, C40, (long)NS * NX, NS, NX, NP, 8, 1.0f, 1.0f);
    k_normalize<<<1, 256, 0, stream>>>(C40, PROTO, CNORM, CFPFH, 0);

    // node features = gammaT x feats / denom   (128x512, K=16384, K-split WMMA)
    k_fill<<<(NB * NS * ND + 255) / 256, 256, 0, stream>>>(NFRAW, 0.0f, NB * NS * ND);
    gemm(stream, GAMT, NP, 1, (long)NS * NP, feats, ND, 1, (long)NP * ND,
         nullptr, 0, NFRAW, (long)NS * ND, NS, ND, NP, 8, 1.0f, 1.0f);
    k_div_rows<<<(NB * NS * ND + 255) / 256, 256, 0, stream>>>(NFRAW, C40, ND);

    // ---- fpfh_gattn on the 128 nodes ----
    k_rownorms<<<1, 256, 0, stream>>>(CFPFH, RN1, NFD);
    k_rownorms<<<1, 256, 0, stream>>>(NFRAW, RN2, ND);
    k_dmat<<<(NB * NS * NS + 255) / 256, 256, 0, stream>>>(CFPFH, RN1, DMAT);
    k_fmat<<<(NB * NS * NS + 255) / 256, 256, 0, stream>>>(NFRAW, RN2, FMAT);
    for (int it = 0; it < RPM_ITERS; ++it) {
        k_rpm_row<<<1, 256, 0, stream>>>(DMAT);
        k_rpm_col<<<1, 256, 0, stream>>>(DMAT);
        k_rpm_row<<<1, 256, 0, stream>>>(FMAT);
        k_rpm_col<<<1, 256, 0, stream>>>(FMAT);
    }
    k_gattn_attn<<<1, 256, 0, stream>>>(DMAT, FMAT);
    // node_feats_out = attn x NFRAW + NFRAW  -> written straight to d_out
    gemm(stream, DMAT, NS, 1, (long)NS * NS, NFRAW, ND, 1, (long)NS * ND,
         NFRAW, (long)NS * ND, outNode, (long)NS * ND, NS, ND, NS, 1, 1.0f, 1.0f);

    // ---- s2p_df ----
    k_s2p_cost<<<(NB * NP + 255) / 256, 256, 0, stream>>>(xyz, PROTO, MMAT);
    k_fill<<<(NB * NP + 255) / 256, 256, 0, stream>>>(UU, 0.0f, NB * NP);
    k_fill<<<1, 256, 0, stream>>>(VV, 0.0f, NB * NS);
    for (int it = 0; it < SK_ITERS; ++it) {
        k_sk_row<<<(NB * NP + 255) / 256, 256, 0, stream>>>(MMAT, VV, UU);
        k_sk_col<<<NB * NS, 256, 0, stream>>>(MMAT, UU, VV);
    }
    k_dscore<<<(int)(((long)NB * NP * NS + 255) / 256), 256, 0, stream>>>(MMAT, UU, VV);
    // f logits = feats x node_featsT / (sqrt(d) * sigma_e)  (WMMA)
    gemm(stream, feats, ND, 1, (long)NP * ND, outNode, 1, ND, (long)NS * ND,
         nullptr, 0, FBUF, (long)NP * NS, NP, NS, ND, 1,
         1.0f / (sqrtf((float)ND) * 0.1f), 1.0f);
    k_row_aux<<<(NB * NP + 255) / 256, 256, 0, stream>>>(FBUF, RAUX);
    k_col_aux<<<NB * NS, 256, 0, stream>>>(FBUF, CAUX);
    k_fs<<<(int)(((long)NB * NP * NS + 255) / 256), 256, 0, stream>>>(FBUF, MMAT, RAUX, CAUX);
    k_rownorm_ip<<<(NB * NP + 255) / 256, 256, 0, stream>>>(FBUF, NS);
    // feats_out = 0.5 * (attn x node_feats + feats)  (WMMA, fused residual)
    gemm(stream, FBUF, NS, 1, (long)NP * NS, outNode, ND, 1, (long)NS * ND,
         feats, (long)NP * ND, outF, (long)NP * ND, NP, ND, NS, 1, 1.0f, 0.5f);

    // ---- geo_decoder ----
    k_top8<<<(NB * NP + 255) / 256, 256, 0, stream>>>(xyz, normals, PROTO, CNORM, W8, I8);
    k_interp<<<(int)(((long)NB * NP * ND + 255) / 256), 256, 0, stream>>>(outF, outNode, W8, I8);
    k_idx_out<<<1, 256, 0, stream>>>(IDX, outIdx);
}
